// EncoderLayer_2138893713471
// MI455X (gfx1250) — matrix operand order repaired and checked by reading, not compiled, not run on hardware
//
#include <hip/hip_runtime.h>
#include <hip/hip_bf16.h>

// ---------------------------------------------------------------------------
// Types for WMMA fragments (gfx1250, wave32)
// ---------------------------------------------------------------------------
typedef __attribute__((ext_vector_type(16))) __bf16 v16bf;
typedef __attribute__((ext_vector_type(8)))  float  v8f;
typedef __attribute__((ext_vector_type(4)))  unsigned int u32x4;

union FragBF16 {
    v16bf bf;
    u32x4 q[2];
};

__device__ inline v8f wmma_bf16(v16bf a, v16bf b, v8f c) {
    // D = A(16x32 bf16) * B(32x16 bf16) + C(16x16 f32)
    return __builtin_amdgcn_wmma_f32_16x16x32_bf16(
        false, a, false, b, (short)0, c, false, false);
}

// A-matrix fragment: elements 0..7 hold k = half*8..+7, elements 8..15 hold
// k = 16+half*8..+7 (16-bit A 16x32 layout) -> two b128 loads.
__device__ inline v16bf load_frag_split(const unsigned short* p0,
                                        const unsigned short* p1) {
    FragBF16 f;
    f.q[0] = *(const u32x4*)p0;
    f.q[1] = *(const u32x4*)p1;
    return f.bf;
}

// Fragment in this code's k order (A and B): element i holds k = half*16 + i, row/col = lane&15 ->
// 16 contiguous elements (32 bytes) when data is laid out [n][k].
__device__ inline v16bf load_frag_contig(const unsigned short* p) {
    FragBF16 f;
    f.q[0] = *(const u32x4*)(p);
    f.q[1] = *(const u32x4*)(p + 8);
    return f.bf;
}

__device__ inline unsigned short f2b(float f) {
    unsigned int u = __float_as_uint(f);
    unsigned int r = u + 0x7FFFu + ((u >> 16) & 1u); // round-to-nearest-even
    return (unsigned short)(r >> 16);
}

// ---------------------------------------------------------------------------
// CDNA5 async global->LDS copy (ASYNCcnt path), 16 bytes per lane.
// ---------------------------------------------------------------------------
__device__ inline void async_b128(void* lds_ptr, const void* gptr) {
    unsigned off = (unsigned)(size_t)lds_ptr;  // low 32 bits = LDS byte addr
    asm volatile("global_load_async_to_lds_b128 %0, %1, off"
                 :: "v"(off), "v"(gptr)
                 : "memory");
}
__device__ inline void wait_async_le4() {
    asm volatile("s_wait_asynccnt 0x4" ::: "memory");
}
__device__ inline void wait_async_0() {
    asm volatile("s_wait_asynccnt 0x0" ::: "memory");
}

// ---------------------------------------------------------------------------
// 16-lane butterfly reductions in pure VALU DPP (no ds_bpermute):
// row_mirror = xor15, row_half_mirror = xor7, quad_perm[3210] = xor3,
// quad_perm[1032] = xor1 -> together cover all 16 lanes.
// ---------------------------------------------------------------------------
__device__ inline float red_max16(float v) {
    int y;
    y = __builtin_amdgcn_mov_dpp(__float_as_int(v), 0x140, 0xF, 0xF, true);
    v = fmaxf(v, __int_as_float(y));
    y = __builtin_amdgcn_mov_dpp(__float_as_int(v), 0x141, 0xF, 0xF, true);
    v = fmaxf(v, __int_as_float(y));
    y = __builtin_amdgcn_mov_dpp(__float_as_int(v), 0x1B, 0xF, 0xF, true);
    v = fmaxf(v, __int_as_float(y));
    y = __builtin_amdgcn_mov_dpp(__float_as_int(v), 0xB1, 0xF, 0xF, true);
    v = fmaxf(v, __int_as_float(y));
    return v;
}
__device__ inline float red_sum16(float v) {
    int y;
    y = __builtin_amdgcn_mov_dpp(__float_as_int(v), 0x140, 0xF, 0xF, true);
    v += __int_as_float(y);
    y = __builtin_amdgcn_mov_dpp(__float_as_int(v), 0x141, 0xF, 0xF, true);
    v += __int_as_float(y);
    y = __builtin_amdgcn_mov_dpp(__float_as_int(v), 0x1B, 0xF, 0xF, true);
    v += __int_as_float(y);
    y = __builtin_amdgcn_mov_dpp(__float_as_int(v), 0xB1, 0xF, 0xF, true);
    v += __int_as_float(y);
    return v;
}

// ---------------------------------------------------------------------------
// fp32 -> bf16 (plain)
// ---------------------------------------------------------------------------
__global__ __launch_bounds__(256) void cvt_f32_bf16(const float* __restrict__ in,
                                                    unsigned short* __restrict__ out,
                                                    int n) {
    int i = blockIdx.x * 256 + threadIdx.x;
    if (i < n) out[i] = f2b(in[i]);
}

// fp32 [K,N] -> bf16 [N,K] transposed (tiled via LDS, both sides coalesced)
__global__ __launch_bounds__(256) void cvt_transpose_bf16(
    const float* __restrict__ in, unsigned short* __restrict__ out, int K, int N) {
    __shared__ unsigned short t[32][33];
    int k0 = blockIdx.y * 32, n0 = blockIdx.x * 32;
    int tx = threadIdx.x & 31, ty = threadIdx.x >> 5;  // 32 x 8
#pragma unroll
    for (int r = 0; r < 32; r += 8)
        t[ty + r][tx] = f2b(in[(size_t)(k0 + ty + r) * N + n0 + tx]);
    __syncthreads();
#pragma unroll
    for (int r = 0; r < 32; r += 8)
        out[(size_t)(n0 + ty + r) * K + k0 + tx] = t[tx][ty + r];
}

// ---------------------------------------------------------------------------
// Tiled bf16 GEMM with WMMA + double-buffered async global->LDS staging.
// C[M,N] = A[M,K] * Bt[N,K]^T + bias.  BM=BN=128, BK=32, 256 threads
// (8 waves; each wave owns a 64x32 sub-tile = 4x2 WMMA accumulators).
// OMODE: 0 = fp32 row-major, 1 = bf16 row-major, 2 = bf16 [B,H,D,S] (for V).
// ---------------------------------------------------------------------------
#define GBM 128
#define GBN 128
#define GBK 32
#define LDK 40   // padded LDS stride (80 bytes: 16B aligned, bank-friendly)

template <bool RELU, int OMODE>
__global__ __launch_bounds__(256) void gemm_bf16_wmma(
    const unsigned short* __restrict__ A, const unsigned short* __restrict__ Bt,
    const float* __restrict__ bias, void* __restrict__ Cout,
    int M, int N, int K) {
    __shared__ unsigned short As[2][GBM * LDK];
    __shared__ unsigned short Bs[2][GBN * LDK];

    const int tid  = threadIdx.x;
    const int lane = tid & 31;
    const int wave = tid >> 5;
    const int l16  = lane & 15;
    const int hlf  = lane >> 4;
    const int wm   = (wave >> 2) * 64;
    const int wn   = (wave & 3) * 32;
    const int m0   = blockIdx.y * GBM;
    const int n0   = blockIdx.x * GBN;

    const int rowA = tid >> 2;          // 0..63, two copies 64 apart
    const int kc   = (tid & 3) * 8;     // 0,8,16,24

    v8f zero = {0.f, 0.f, 0.f, 0.f, 0.f, 0.f, 0.f, 0.f};
    v8f acc[4][2];
#pragma unroll
    for (int mt = 0; mt < 4; ++mt)
#pragma unroll
        for (int nt = 0; nt < 2; ++nt) acc[mt][nt] = zero;

    // 4 async b128 per wave per tile: 2 for A, 2 for B
    auto issue_tile = [&](int buf, int kb) {
#pragma unroll
        for (int r = 0; r < 2; ++r) {
            int row = rowA + r * 64;
            async_b128(&As[buf][row * LDK + kc],
                       &A[(size_t)(m0 + row) * K + kb + kc]);
            async_b128(&Bs[buf][row * LDK + kc],
                       &Bt[(size_t)(n0 + row) * K + kb + kc]);
        }
    };

    const int nk = K / GBK;
    issue_tile(0, 0);

    for (int t = 0; t < nk; ++t) {
        const int cur = t & 1;
        if (t + 1 < nk) {
            issue_tile(cur ^ 1, (t + 1) * GBK);
            wait_async_le4();   // tile t done (async loads complete in order)
        } else {
            wait_async_0();
        }
        __syncthreads();

        v16bf af[4], bfr[2];
#pragma unroll
        for (int mt = 0; mt < 4; ++mt) {
            const unsigned short* p = &As[cur][(wm + mt * 16 + l16) * LDK + hlf * 16];
            af[mt] = load_frag_contig(p);
        }
#pragma unroll
        for (int nt = 0; nt < 2; ++nt)
            bfr[nt] = load_frag_contig(&Bs[cur][(wn + nt * 16 + l16) * LDK + hlf * 16]);

#pragma unroll
        for (int mt = 0; mt < 4; ++mt)
#pragma unroll
            for (int nt = 0; nt < 2; ++nt)
                acc[mt][nt] = wmma_bf16(af[mt], bfr[nt], acc[mt][nt]);
        __syncthreads();  // protect buffer `cur` before it is refilled
    }

    // Epilogue: C layout — VGPR j holds row (j + half*8), col = lane&15
#pragma unroll
    for (int mt = 0; mt < 4; ++mt)
#pragma unroll
        for (int nt = 0; nt < 2; ++nt) {
            int col  = n0 + wn + nt * 16 + l16;
            float bv = bias[col];
#pragma unroll
            for (int j = 0; j < 8; ++j) {
                int row = m0 + wm + mt * 16 + hlf * 8 + j;
                float v = acc[mt][nt][j] + bv;
                if (RELU) v = fmaxf(v, 0.f);
                if constexpr (OMODE == 0) {
                    ((float*)Cout)[(size_t)row * N + col] = v;
                } else if constexpr (OMODE == 1) {
                    ((unsigned short*)Cout)[(size_t)row * N + col] = f2b(v);
                } else {
                    // V-transposed: [B,H,D,S]; row = b*2048+s, col = h*64+d
                    int b = row >> 11, s = row & 2047;
                    ((unsigned short*)Cout)[((size_t)b * 1024 + col) * 2048 + s] = f2b(v);
                }
            }
        }
}

// ---------------------------------------------------------------------------
// Flash attention: 1 wave per (b, h, 16-query tile).
// Q,K: [B,S,H,64] bf16;  V: [B,H,D,S] bf16 (pre-transposed by the V GEMM).
// Scores = Q.K^T / 8 with online softmax over S in 32-key chunks.
// ---------------------------------------------------------------------------
__global__ __launch_bounds__(32) void flash_attn_wmma(
    const unsigned short* __restrict__ Qg, const unsigned short* __restrict__ Kg,
    const unsigned short* __restrict__ Vg, const int* __restrict__ mask,
    unsigned short* __restrict__ Og) {
    const int S = 2048, H = 16;
    __shared__ unsigned short Ps[16 * LDK];  // P tile (16 q x 32 k) bf16

    const int lane = threadIdx.x;
    const int l16  = lane & 15;
    const int hlf  = lane >> 4;
    const int qb   = blockIdx.x & 127;        // S/16 = 128
    const int h    = (blockIdx.x >> 7) & 15;
    const int b    = blockIdx.x >> 11;

    // Q fragment (A-matrix): row = qb*16 + l16, contiguous over d
    const unsigned short* qp = Qg + ((size_t)(b * S + qb * 16 + l16) * H + h) * 64;
    v16bf qf[2];
#pragma unroll
    for (int ds = 0; ds < 2; ++ds) {
        const unsigned short* p = qp + ds * 32 + hlf * 16;
        qf[ds] = load_frag_contig(p);
    }

    v8f zero = {0.f, 0.f, 0.f, 0.f, 0.f, 0.f, 0.f, 0.f};
    v8f accO[4];
    float m_i[8], l_i[8];
#pragma unroll
    for (int nt = 0; nt < 4; ++nt) accO[nt] = zero;
#pragma unroll
    for (int j = 0; j < 8; ++j) { m_i[j] = -3.0e38f; l_i[j] = 0.f; }

    // V base for this (b,h): Vg[((b*16+h)*64 + d)*2048 + s]
    const unsigned short* vbase = Vg + ((size_t)(b * 16 + h) * 64) * 2048;

    for (int kv0 = 0; kv0 < S; kv0 += 32) {
        // K fragments (B-matrix of Q.K^T): col = key, k = d (contiguous)
        v16bf kf[2][2];
#pragma unroll
        for (int t = 0; t < 2; ++t) {
            const unsigned short* kp =
                Kg + ((size_t)(b * S + kv0 + t * 16 + l16) * H + h) * 64;
#pragma unroll
            for (int ds = 0; ds < 2; ++ds)
                kf[t][ds] = load_frag_contig(kp + ds * 32 + hlf * 16);
        }
        // V fragments (B-matrix of P.V): col = d, k = key (contiguous in [*,S])
        v16bf vf[4];
#pragma unroll
        for (int nt = 0; nt < 4; ++nt)
            vf[nt] = load_frag_contig(
                vbase + (size_t)(nt * 16 + l16) * 2048 + kv0 + hlf * 16);

        v8f s0 = zero, s1 = zero;
        s0 = wmma_bf16(qf[0], kf[0][0], s0);
        s0 = wmma_bf16(qf[1], kf[0][1], s0);
        s1 = wmma_bf16(qf[0], kf[1][0], s1);
        s1 = wmma_bf16(qf[1], kf[1][1], s1);

        int mv0 = mask[b * S + kv0 + l16];
        int mv1 = mask[b * S + kv0 + 16 + l16];

        float mnew[8];
#pragma unroll
        for (int j = 0; j < 8; ++j) {
            float a0 = mv0 ? s0[j] * 0.125f : -1.0e9f;
            float a1 = mv1 ? s1[j] * 0.125f : -1.0e9f;
            s0[j] = a0; s1[j] = a1;
            mnew[j] = fmaxf(m_i[j], red_max16(fmaxf(a0, a1)));
        }
#pragma unroll
        for (int j = 0; j < 8; ++j) {
            float sc = __expf(m_i[j] - mnew[j]);
            m_i[j] = mnew[j];
            float p0 = __expf(s0[j] - mnew[j]);
            float p1 = __expf(s1[j] - mnew[j]);
            l_i[j] = l_i[j] * sc + red_sum16(p0 + p1);
#pragma unroll
            for (int nt = 0; nt < 4; ++nt) accO[nt][j] *= sc;
            int prow = hlf * 8 + j;
            Ps[prow * LDK + l16]      = f2b(p0);
            Ps[prow * LDK + 16 + l16] = f2b(p1);
        }
        __syncthreads();

        const unsigned short* pp = &Ps[l16 * LDK + hlf * 16];
        v16bf pa = load_frag_contig(pp);
#pragma unroll
        for (int nt = 0; nt < 4; ++nt)
            accO[nt] = wmma_bf16(pa, vf[nt], accO[nt]);
        __syncthreads();
    }

#pragma unroll
    for (int j = 0; j < 8; ++j) {
        float inv = 1.f / l_i[j];
        int row = qb * 16 + hlf * 8 + j;
        size_t base = ((size_t)(b * S + row) * H + h) * 64;
#pragma unroll
        for (int nt = 0; nt < 4; ++nt)
            Og[base + nt * 16 + l16] = f2b(accO[nt][j] * inv);
    }
}

// ---------------------------------------------------------------------------
// Residual add + LayerNorm over D=1024.  One block (256 thr) per row.
// ---------------------------------------------------------------------------
__global__ __launch_bounds__(256) void residual_ln(
    const float* __restrict__ xa, const float* __restrict__ xr,
    const float* __restrict__ g, const float* __restrict__ be,
    float* __restrict__ out, unsigned short* __restrict__ out_bf) {
    const int D = 1024;
    int row = blockIdx.x;
    int tid = threadIdx.x;
    __shared__ float red[256];

    float t[4];
    float s = 0.f;
#pragma unroll
    for (int i = 0; i < 4; ++i) {
        int c = tid + i * 256;
        float v = xa[(size_t)row * D + c] + xr[(size_t)row * D + c];
        t[i] = v;
        s += v;
    }
    red[tid] = s;
    __syncthreads();
    for (int off = 128; off > 0; off >>= 1) {
        if (tid < off) red[tid] += red[tid + off];
        __syncthreads();
    }
    float mu = red[0] * (1.f / D);
    __syncthreads();

    float s2 = 0.f;
#pragma unroll
    for (int i = 0; i < 4; ++i) {
        float d = t[i] - mu;
        s2 += d * d;
    }
    red[tid] = s2;
    __syncthreads();
    for (int off = 128; off > 0; off >>= 1) {
        if (tid < off) red[tid] += red[tid + off];
        __syncthreads();
    }
    float inv = rsqrtf(red[0] * (1.f / D) + 1e-5f);

#pragma unroll
    for (int i = 0; i < 4; ++i) {
        int c = tid + i * 256;
        float y = (t[i] - mu) * inv * g[c] + be[c];
        out[(size_t)row * D + c] = y;
        if (out_bf) out_bf[(size_t)row * D + c] = f2b(y);
    }
}

// ---------------------------------------------------------------------------
// Launcher
// ---------------------------------------------------------------------------
extern "C" void kernel_launch(void* const* d_in, const int* in_sizes, int n_in,
                              void* d_out, int out_size, void* d_ws, size_t ws_size,
                              hipStream_t stream) {
    (void)in_sizes; (void)n_in; (void)out_size; (void)ws_size;
    const int B = 4, S = 2048, D = 1024, FF = 4096;
    const int M = B * S;  // 8192

    const float* x     = (const float*)d_in[0];
    const int*   mask  = (const int*)  d_in[1];
    const float* w_q   = (const float*)d_in[2];
    const float* b_q   = (const float*)d_in[3];
    const float* w_k   = (const float*)d_in[4];
    const float* b_k   = (const float*)d_in[5];
    const float* w_v   = (const float*)d_in[6];
    const float* b_v   = (const float*)d_in[7];
    const float* w_o   = (const float*)d_in[8];
    const float* b_o   = (const float*)d_in[9];
    const float* w_ff1 = (const float*)d_in[10];
    const float* b_ff1 = (const float*)d_in[11];
    const float* w_ff2 = (const float*)d_in[12];
    const float* b_ff2 = (const float*)d_in[13];
    const float* ln1_g = (const float*)d_in[14];
    const float* ln1_b = (const float*)d_in[15];
    const float* ln2_g = (const float*)d_in[16];
    const float* ln2_b = (const float*)d_in[17];

    const size_t MB = 1024u * 1024u;
    char* ws = (char*)d_ws;
    unsigned short* xb   = (unsigned short*)(ws + 0);        // 16 MB
    unsigned short* wqb  = (unsigned short*)(ws + 16 * MB);  //  2 MB  [N,K]
    unsigned short* wkb  = (unsigned short*)(ws + 18 * MB);
    unsigned short* wvb  = (unsigned short*)(ws + 20 * MB);
    unsigned short* wob  = (unsigned short*)(ws + 22 * MB);
    unsigned short* wf1b = (unsigned short*)(ws + 24 * MB);  //  8 MB  [FF,D]
    unsigned short* wf2b = (unsigned short*)(ws + 32 * MB);  //  8 MB  [D,FF]
    unsigned short* qb   = (unsigned short*)(ws + 40 * MB);  // 16 MB
    unsigned short* kb   = (unsigned short*)(ws + 56 * MB);  // 16 MB
    unsigned short* vb   = (unsigned short*)(ws + 72 * MB);  // 16 MB [B,H,D,S]
    unsigned short* ab   = (unsigned short*)(ws + 88 * MB);  // 16 MB
    unsigned short* hb   = (unsigned short*)(ws + 40 * MB);  // reuse (64 MB)
    float*          ptmp = (float*)(ws + 104 * MB);          // 32 MB (reused)
    float*          x1   = (float*)(ws + 136 * MB);          // 32 MB
    unsigned short* x1b  = (unsigned short*)(ws + 0);        // reuse xb

    // 1) activations bf16; weights bf16 *transposed* to [N,K]
    cvt_f32_bf16<<<(M * D + 255) / 256, 256, 0, stream>>>(x, xb, M * D);
    auto cvtT = [&](const float* src, unsigned short* dst, int K, int N) {
        cvt_transpose_bf16<<<dim3(N / 32, K / 32), 256, 0, stream>>>(src, dst, K, N);
    };
    cvtT(w_q, wqb, D, D);
    cvtT(w_k, wkb, D, D);
    cvtT(w_v, wvb, D, D);
    cvtT(w_o, wob, D, D);
    cvtT(w_ff1, wf1b, D, FF);
    cvtT(w_ff2, wf2b, FF, D);

    // 2) QKV projections (V written transposed [B,H,D,S])
    dim3 gDD(D / GBN, M / GBM);
    gemm_bf16_wmma<false, 1><<<gDD, 256, 0, stream>>>(xb, wqb, b_q, qb, M, D, D);
    gemm_bf16_wmma<false, 1><<<gDD, 256, 0, stream>>>(xb, wkb, b_k, kb, M, D, D);
    gemm_bf16_wmma<false, 2><<<gDD, 256, 0, stream>>>(xb, wvb, b_v, vb, M, D, D);

    // 3) flash attention -> bf16 [B,S,H,64] (== [M, D] row-major)
    flash_attn_wmma<<<B * 16 * (S / 16), 32, 0, stream>>>(qb, kb, vb, mask, ab);

    // 4) output projection -> fp32
    gemm_bf16_wmma<false, 0><<<gDD, 256, 0, stream>>>(ab, wob, b_o, ptmp, M, D, D);

    // 5) x1 = LN(x + attn_out)  (fp32 + bf16 copy)
    residual_ln<<<M, 256, 0, stream>>>(x, ptmp, ln1_g, ln1_b, x1, x1b);

    // 6) FFN up + ReLU -> bf16
    dim3 gDF(FF / GBN, M / GBM);
    gemm_bf16_wmma<true, 1><<<gDF, 256, 0, stream>>>(x1b, wf1b, b_ff1, hb, M, FF, D);

    // 7) FFN down -> fp32 (reuse ptmp)
    gemm_bf16_wmma<false, 0><<<gDD, 256, 0, stream>>>(hb, wf2b, b_ff2, ptmp, M, D, FF);

    // 8) out = LN(x1 + ff)
    residual_ln<<<M, 256, 0, stream>>>(x1, ptmp, ln2_g, ln2_b, (float*)d_out, nullptr);
}
